// LongMultiheadedAttention_55740085567719
// MI455X (gfx1250) — matrix operand order repaired and checked by reading, not compiled, not run on hardware
//
#include <hip/hip_runtime.h>
#include <math.h>
#include <stdint.h>

// Problem constants (match reference setup_inputs)
#define B_  2
#define T_  4096
#define H_  16
#define S_  64      // head dim
#define D_  1024    // model dim
#define W_  128     // one-sided window
#define CH  64      // query chunk per block
#define KW  (CH + 2 * W_)   // 320 keys per window

typedef __attribute__((ext_vector_type(16))) __bf16 v16bf;
typedef __attribute__((ext_vector_type(8)))  __bf16 bf16x8;
typedef __attribute__((ext_vector_type(8)))  float  v8f;

// ---------------------------------------------------------------------------
// CDNA5 async global->LDS copy (ASYNCcnt path, ISA §15.18.3 op 98).
// vdst = LDS byte offset, vaddr = 64-bit global address, saddr = off.
// ---------------------------------------------------------------------------
__device__ __forceinline__ void async_copy_b128(uint32_t lds_off, const void* g) {
  asm volatile("global_load_async_to_lds_b128 %0, %1, off"
               :: "v"(lds_off), "v"((uint64_t)(uintptr_t)g) : "memory");
}
__device__ __forceinline__ void wait_async0() {
  asm volatile("s_wait_asynccnt 0" ::: "memory");
}
__device__ __forceinline__ uint32_t lds_off32(const void* p) {
  return (uint32_t)(uintptr_t)p;   // generic LDS addr: low 32 bits = LDS offset
}

// ---------------------------------------------------------------------------
// WMMA helpers: v_wmma_f32_16x16x32_bf16, wave32 layouts per CDNA5 ISA 7.12.2
// ---------------------------------------------------------------------------
__device__ __forceinline__ v8f wmma_bf16(v16bf a, v16bf b, v8f c) {
  return __builtin_amdgcn_wmma_f32_16x16x32_bf16(
      /*neg_a=*/false, a, /*neg_b=*/false, b,
      /*c_mod=*/(short)0, c, /*reuse_a=*/false, /*reuse_b=*/false);
}

__device__ __forceinline__ v16bf cat8(bf16x8 lo, bf16x8 hi) {
  return __builtin_shufflevector(lo, hi, 0, 1, 2, 3, 4, 5, 6, 7,
                                 8, 9, 10, 11, 12, 13, 14, 15);
}

// A fragment: 16x32 (MxK), source row-major A[M][K], row stride ld (bf16).
// M = lane&15; elems 0..7 -> K = khi+0..7, elems 8..15 -> K = khi+16..23,
// khi = 8*(lane>=16).  Two ds_load_b128 per fragment.
__device__ __forceinline__ v16bf load_A16x32(const __bf16* base, int ld, int lane) {
  const int m   = lane & 15;
  const int khi = (lane >> 4) << 3;
  const __bf16* p = base + m * ld + khi;
  bf16x8 lo = *(const bf16x8*)(p);
  bf16x8 hi = *(const bf16x8*)(p + 16);
  return cat8(lo, hi);
}

// B fragment: 32x16 (KxN) from a TRANSPOSED source src[n][k] row-major with
// row stride ld.  N = lane&15; elems 0..7 -> K = k0+0..7, elems 8..15 -> K = k0+16..23, k0 = 8*(lane>=16).
__device__ __forceinline__ v16bf load_B32x16_T(const __bf16* base, int ld, int lane) {
  const int n  = lane & 15;
  const int k0 = (lane >> 4) << 3;
  const __bf16* p = base + n * ld + k0;
  bf16x8 lo = *(const bf16x8*)(p);
  bf16x8 hi = *(const bf16x8*)(p + 16);
  return cat8(lo, hi);
}

// ---------------------------------------------------------------------------
// Kernel 1: per-head QKV projection.  q = (x@Wq^T)/8, k = x@Wk^T, v = x@Wv^T
// Output layout: [b*h][t][64] bf16.  B = W^T so the [N][K] LDS tile is W
// row-major (straight copy).
// ---------------------------------------------------------------------------
__global__ __launch_bounds__(128)
void qkv_proj_kernel(const float* __restrict__ x,
                     const float* __restrict__ Wq,
                     const float* __restrict__ Wk,
                     const float* __restrict__ Wv,
                     __bf16* __restrict__ qbuf,
                     __bf16* __restrict__ kbuf,
                     __bf16* __restrict__ vbuf) {
  __shared__ __align__(16) __bf16 xs[CH * S_];       // 8 KB
  __shared__ __align__(16) __bf16 wl[3 * S_ * S_];   // 24 KB, W row-major

  const int tid  = threadIdx.x;
  const int lane = tid & 31;
  const int wave = tid >> 5;
  const int chunk = blockIdx.x % (T_ / CH);
  const int bh    = blockIdx.x / (T_ / CH);
  const int bb    = bh / H_;
  const int hh    = bh % H_;
  const int t0    = chunk * CH;

  for (int i = tid; i < CH * S_; i += 128) {
    const int r = i >> 6, c = i & 63;
    xs[i] = (__bf16)x[((size_t)bb * T_ + t0 + r) * D_ + hh * S_ + c];
  }
  const float* Ws[3] = {Wq, Wk, Wv};
  for (int i = tid; i < 3 * S_ * S_; i += 128)
    wl[i] = (__bf16)Ws[i >> 12][i & 4095];   // straight copy: W[o][i]
  __syncthreads();

  const int m0 = wave * 16;
  // A fragments depend only on kk: load once, reuse for all 12 output tiles.
  const v16bf a0 = load_A16x32(&xs[m0 * S_ + 0],  S_, lane);
  const v16bf a1 = load_A16x32(&xs[m0 * S_ + 32], S_, lane);

  __bf16* outp[3] = {qbuf, kbuf, vbuf};
#pragma unroll
  for (int mat = 0; mat < 3; ++mat) {
    const float scale = (mat == 0) ? 0.125f : 1.0f;  // 1/sqrt(64) on q
#pragma unroll
    for (int nt = 0; nt < 4; ++nt) {
      const __bf16* bbase = &wl[mat * S_ * S_ + (nt * 16) * S_];
      v8f acc = {};
      acc = wmma_bf16(a0, load_B32x16_T(bbase + 0,  S_, lane), acc);
      acc = wmma_bf16(a1, load_B32x16_T(bbase + 32, S_, lane), acc);
      const int col = nt * 16 + (lane & 15);
#pragma unroll
      for (int r = 0; r < 8; ++r) {
        const int row = m0 + r + ((lane >> 4) << 3);
        outp[mat][((size_t)bh * T_ + t0 + row) * S_ + col] = (__bf16)(acc[r] * scale);
      }
    }
  }
}

// ---------------------------------------------------------------------------
// Kernel 2: sliding-window attention for one (b,h,64-query chunk).
// Keys/values window = [t0-128, t0+64+128) -> 320 keys (zero-padded OOB).
// Band rule: query row r attends to window cols j in [r, r+256].
// ---------------------------------------------------------------------------
__global__ __launch_bounds__(128)
void sliding_attn_kernel(const __bf16* __restrict__ qbuf,
                         const __bf16* __restrict__ kbuf,
                         const __bf16* __restrict__ vbuf,
                         const int* __restrict__ mask,
                         __bf16* __restrict__ attn) {
  extern __shared__ __align__(16) char smem_raw[];
  __bf16* qtile  = (__bf16*)smem_raw;               // CH*S_  bf16  (8 KB)
  __bf16* kwin   = qtile + CH * S_;                 // KW*S_  bf16  (40 KB) [key][dim]
  __bf16* vwinT  = kwin + KW * S_;                  // S_*KW  bf16  (40 KB) [dim][key]
  float*  scores = (float*)(vwinT + S_ * KW);       // CH*KW  f32   (80 KB)
  __bf16* probs  = (__bf16*)(scores + CH * KW);     // CH*KW  bf16  (40 KB)
  float*  red    = (float*)(probs + CH * KW);       // 128    f32

  const int tid  = threadIdx.x;
  const int lane = tid & 31;
  const int wave = tid >> 5;
  const int chunk = blockIdx.x % (T_ / CH);
  const int bh    = blockIdx.x / (T_ / CH);
  const int bb    = bh / H_;
  const int hh    = bh % H_;
  const int t0    = chunk * CH;
  const int k0g   = t0 - W_;      // global key pos of window col 0

  const bool interior = (k0g >= 0) && (k0g + KW <= T_);
  if (interior) {
    // Q tile (8 KB) and K window (40 KB): contiguous copies -> async b128.
    const char* qsrc = (const char*)(qbuf + ((size_t)bh * T_ + t0) * S_);
    const char* ksrc = (const char*)(kbuf + ((size_t)bh * T_ + k0g) * S_);
    const uint32_t qdst = lds_off32(qtile);
    const uint32_t kdst = lds_off32(kwin);
#pragma unroll
    for (int it = 0; it < 4; ++it) {                 // 4*128*16B = 8 KB
      const int o = (tid + it * 128) * 16;
      async_copy_b128(qdst + o, qsrc + o);
    }
#pragma unroll
    for (int it = 0; it < 20; ++it) {                // 20*128*16B = 40 KB
      const int o = (tid + it * 128) * 16;
      async_copy_b128(kdst + o, ksrc + o);
    }
    // V needs a transpose -> scalar staging (overlaps with async K/Q).
    for (int i = tid; i < KW * S_; i += 128) {
      const int j = i >> 6, c = i & 63;
      vwinT[c * KW + j] = vbuf[((size_t)bh * T_ + k0g + j) * S_ + c];
    }
    wait_async0();
  } else {
    for (int i = tid; i < CH * S_; i += 128)
      qtile[i] = qbuf[((size_t)bh * T_ + t0) * S_ + i];
    for (int i = tid; i < KW * S_; i += 128) {
      const int j = i >> 6, c = i & 63;
      const int kpos = k0g + j;
      const bool ok = (kpos >= 0) && (kpos < T_);
      const __bf16 kv = ok ? kbuf[((size_t)bh * T_ + kpos) * S_ + c] : (__bf16)0.0f;
      const __bf16 vv = ok ? vbuf[((size_t)bh * T_ + kpos) * S_ + c] : (__bf16)0.0f;
      kwin[j * S_ + c]  = kv;
      vwinT[c * KW + j] = vv;
    }
  }
  __syncthreads();

  // ---- scores = Q @ K^T : A fragments hoisted, 20 key tiles per wave ----
  const int m0 = wave * 16;
  const v16bf qa0 = load_A16x32(&qtile[m0 * S_ + 0],  S_, lane);
  const v16bf qa1 = load_A16x32(&qtile[m0 * S_ + 32], S_, lane);
  for (int nt = 0; nt < KW / 16; ++nt) {
    const __bf16* bbase = &kwin[(nt * 16) * S_];
    v8f acc = {};
    acc = wmma_bf16(qa0, load_B32x16_T(bbase + 0,  S_, lane), acc);
    acc = wmma_bf16(qa1, load_B32x16_T(bbase + 32, S_, lane), acc);
    const int col = nt * 16 + (lane & 15);
#pragma unroll
    for (int r = 0; r < 8; ++r) {
      const int row = m0 + r + ((lane >> 4) << 3);
      scores[row * KW + col] = acc[r];
    }
  }
  __syncthreads();

  // ---- band mask + key mask + softmax: 2 threads per row, 160 cols each ----
  {
    const int row  = tid >> 1;
    const int j0   = (tid & 1) * (KW / 2);
    float* sr = scores + row * KW;
    float pm = -1e30f;
    for (int j = j0; j < j0 + KW / 2; ++j) {
      const int kpos = k0g + j;
      const bool valid = (j >= row) && (j <= row + 2 * W_) &&
                         (kpos >= 0) && (kpos < T_);
      float s = -1e30f;
      if (valid) {
        s = sr[j];
        if (mask[bb * T_ + kpos] != 0) s -= 10000.0f;
      }
      sr[j] = s;
      pm = fmaxf(pm, s);
    }
    red[tid] = pm;
    __syncthreads();
    const float mx = fmaxf(red[row * 2], red[row * 2 + 1]);
    __syncthreads();
    float ps = 0.0f;
    for (int j = j0; j < j0 + KW / 2; ++j) {
      const float e = __expf(sr[j] - mx);   // invalid -> exp(-huge) == 0
      ps += e;
      sr[j] = e;
    }
    red[tid] = ps;
    __syncthreads();
    const float inv = 1.0f / (red[row * 2] + red[row * 2 + 1]);
    for (int j = j0; j < j0 + KW / 2; ++j)
      probs[row * KW + j] = (__bf16)(sr[j] * inv);
  }
  __syncthreads();

  // ---- out = P @ V : k-outer loop, A loaded once per k-step, 4 acc tiles ----
  v8f acc[4] = {};
#pragma unroll
  for (int kk = 0; kk < KW; kk += 32) {
    const v16bf a = load_A16x32(&probs[m0 * KW + kk], KW, lane);
#pragma unroll
    for (int nt = 0; nt < 4; ++nt) {
      const v16bf b = load_B32x16_T(&vwinT[(nt * 16) * KW + kk], KW, lane);
      acc[nt] = wmma_bf16(a, b, acc[nt]);
    }
  }
#pragma unroll
  for (int nt = 0; nt < 4; ++nt) {
    const int col = hh * S_ + nt * 16 + (lane & 15);
#pragma unroll
    for (int r = 0; r < 8; ++r) {
      const int row = t0 + m0 + r + ((lane >> 4) << 3);
      attn[((size_t)bb * T_ + row) * D_ + col] = (__bf16)acc[nt][r];
    }
  }
}

// ---------------------------------------------------------------------------
// Kernel 3: out = attn @ Wo^T + bo  (M=8192, N=K=1024), 64x64 tiles.
// B = Wo^T so the [N][K] tile is Wo row-major, staged straight from f32 Wo.
// A tile (bf16, contiguous 128B rows) staged with async b128 copies.
// ---------------------------------------------------------------------------
__global__ __launch_bounds__(128)
void out_proj_kernel(const __bf16* __restrict__ attn,
                     const float* __restrict__ Wo,
                     const float* __restrict__ bo,
                     float* __restrict__ out) {
  __shared__ __align__(16) __bf16 atile[64 * 64];    // 8 KB  [m][k]
  __shared__ __align__(16) __bf16 btileT[64 * 64];   // 8 KB  [n][k]

  const int tid  = threadIdx.x;
  const int lane = tid & 31;
  const int wave = tid >> 5;
  const int nblk = blockIdx.x % (D_ / 64);
  const int mblk = blockIdx.x / (D_ / 64);
  const int m0g  = mblk * 64, n0g = nblk * 64;
  const int mw   = wave * 16;
  const uint32_t adst = lds_off32(atile);

  v8f acc[4] = {};
  for (int k0 = 0; k0 < D_; k0 += 64) {
    // A tile: 64 rows x 128 B -> 8 async b128 transfers per row.
#pragma unroll
    for (int it = 0; it < 4; ++it) {
      const int i   = tid + it * 128;      // 0..511
      const int r   = i >> 3, seg = (i & 7) * 8;
      async_copy_b128(adst + (uint32_t)(r * 64 + seg) * 2,
                      attn + (size_t)(m0g + r) * D_ + k0 + seg);
    }
    // B tile: f32 -> bf16 conversion, scalar staging.
    for (int i = tid; i < 64 * 64; i += 128) {
      const int r = i >> 6, c = i & 63;
      btileT[i] = (__bf16)Wo[(size_t)(n0g + r) * D_ + k0 + c];
    }
    if (k0 + 64 < D_) {
      __builtin_prefetch(&attn[(size_t)(m0g + (tid >> 1)) * D_ + k0 + 64], 0, 1);
      __builtin_prefetch(&Wo[(size_t)(n0g + (tid >> 1)) * D_ + k0 + 64], 0, 1);
    }
    wait_async0();
    __syncthreads();
#pragma unroll
    for (int kk = 0; kk < 64; kk += 32) {
      const v16bf a = load_A16x32(&atile[mw * 64 + kk], 64, lane);
#pragma unroll
      for (int nt = 0; nt < 4; ++nt) {
        const v16bf b = load_B32x16_T(&btileT[(nt * 16) * 64 + kk], 64, lane);
        acc[nt] = wmma_bf16(a, b, acc[nt]);
      }
    }
    __syncthreads();
  }
#pragma unroll
  for (int nt = 0; nt < 4; ++nt) {
    const int col = n0g + nt * 16 + (lane & 15);
    const float bias = bo[col];
#pragma unroll
    for (int r = 0; r < 8; ++r) {
      const int row = m0g + mw + r + ((lane >> 4) << 3);
      out[(size_t)row * D_ + col] = acc[nt][r] + bias;
    }
  }
}

// ---------------------------------------------------------------------------
// Launch
// ---------------------------------------------------------------------------
extern "C" void kernel_launch(void* const* d_in, const int* in_sizes, int n_in,
                              void* d_out, int out_size, void* d_ws, size_t ws_size,
                              hipStream_t stream) {
  (void)in_sizes; (void)n_in; (void)out_size; (void)ws_size;
  const float* x    = (const float*)d_in[0];
  const int*   mask = (const int*)d_in[1];
  const float* Wq   = (const float*)d_in[2];
  const float* Wk   = (const float*)d_in[3];
  const float* Wv   = (const float*)d_in[4];
  const float* Wo   = (const float*)d_in[5];
  const float* bo   = (const float*)d_in[6];
  float* out = (float*)d_out;

  // Workspace: q,k,v [32][4096][64] bf16 (16 MB each) + attn [2][4096][1024]
  // bf16 (16 MB)  ->  64 MB total.
  const size_t per = (size_t)B_ * H_ * T_ * S_;
  __bf16* qbuf = (__bf16*)d_ws;
  __bf16* kbuf = qbuf + per;
  __bf16* vbuf = kbuf + per;
  __bf16* attn = vbuf + per;

  const dim3 blk(128);
  const dim3 grid_bh(B_ * H_ * (T_ / CH));   // 2048 blocks

  qkv_proj_kernel<<<grid_bh, blk, 0, stream>>>(x, Wq, Wk, Wv, qbuf, kbuf, vbuf);

  const size_t lds2 = (size_t)(CH * S_ + 2 * KW * S_) * sizeof(__bf16) +  // q + k + vT
                      (size_t)CH * KW * sizeof(float) +                    // scores
                      (size_t)CH * KW * sizeof(__bf16) +                   // probs
                      128 * sizeof(float);                                 // reductions
  sliding_attn_kernel<<<grid_bh, blk, lds2, stream>>>(qbuf, kbuf, vbuf, mask, attn);

  const dim3 grid_o((B_ * T_ / 64) * (D_ / 64));   // 2048 blocks
  out_proj_kernel<<<grid_o, blk, 0, stream>>>(attn, Wo, bo, out);
}
